// FuzzySelfAttention_31542239822482
// MI455X (gfx1250) — compile-verified, matrix operand order checked by reading, not run on hardware
//
#include <hip/hip_runtime.h>
#include <stdint.h>

// ---------------------------------------------------------------------------
// CDNA5 bf16 WMMA types / helpers
// ---------------------------------------------------------------------------
typedef __bf16 bf16_t;
typedef bf16_t v16bf __attribute__((ext_vector_type(16)));
typedef float  v8f   __attribute__((ext_vector_type(8)));

union Frag {
  v16bf    v;
  uint32_t u[8];
};

__device__ __forceinline__ uint16_t f2bf(float f) {
  uint32_t u = __float_as_uint(f);
  uint32_t r = u + 0x7FFFu + ((u >> 16) & 1u);   // round-to-nearest-even
  return (uint16_t)(r >> 16);
}
__device__ __forceinline__ float bf2f(uint16_t h) {
  return __uint_as_float(((uint32_t)h) << 16);
}

// Load one 16x32 bf16 WMMA fragment (A or B operand in "NT" K-contiguous
// form). p0 points at element (row0, k0) of a row-major [*, ld] bf16 matrix.
// ISA 16-bit 16x32 layout: lanes 0-15 row = lane, K = {base+0..7, base+16..23};
// lanes 16-31 row = lane-16, K = {base+8..15, base+24..31}. Two b128 loads.
__device__ __forceinline__ v16bf load_frag(const uint16_t* p0, int ld, int lane) {
  const uint16_t* p = p0 + (size_t)(lane & 15) * ld + ((lane >> 4) << 3);
  uint4 a = *reinterpret_cast<const uint4*>(p);
  uint4 b = *reinterpret_cast<const uint4*>(p + 16);
  Frag f;
  f.u[0] = a.x; f.u[1] = a.y; f.u[2] = a.z; f.u[3] = a.w;
  f.u[4] = b.x; f.u[5] = b.y; f.u[6] = b.z; f.u[7] = b.w;
  return f.v;
}

__device__ __forceinline__ v8f wmma_bf16(v16bf a, v16bf b, v8f c) {
  // D = A * B + C, 16x16x32, f32 accumulate
  return __builtin_amdgcn_wmma_f32_16x16x32_bf16(false, a, false, b,
                                                 (short)0, c, false, false);
}

// ---------------------------------------------------------------------------
// 1) fp32 -> bf16 conversion (8 elements / thread, 16B loads+stores)
// ---------------------------------------------------------------------------
__global__ __launch_bounds__(256) void cvt_f32_bf16(
    const float* __restrict__ in, uint16_t* __restrict__ out, int n8) {
  int i = blockIdx.x * 256 + threadIdx.x;
  if (i >= n8) return;
  const float4* p = reinterpret_cast<const float4*>(in) + (size_t)i * 2;
  float4 a = p[0], b = p[1];
  uint4 o;
  o.x = (uint32_t)f2bf(a.x) | ((uint32_t)f2bf(a.y) << 16);
  o.y = (uint32_t)f2bf(a.z) | ((uint32_t)f2bf(a.w) << 16);
  o.z = (uint32_t)f2bf(b.x) | ((uint32_t)f2bf(b.y) << 16);
  o.w = (uint32_t)f2bf(b.z) | ((uint32_t)f2bf(b.w) << 16);
  reinterpret_cast<uint4*>(out)[i] = o;
}

// ---------------------------------------------------------------------------
// 2) C[m,n] = sum_k A[m,k]*B[n,k] + bias[n]   (QKV projections)
//    A: [M x Kd] bf16, B: [N x Kd] bf16, C: [M x N] bf16.
//    128 threads = 4 waves; block tile 32(M) x 256(N); wave tile 32x64
//    (2x4 sub-tiles -> 8 WMMAs per 6 fragment loads per K-step).
// ---------------------------------------------------------------------------
__global__ __launch_bounds__(128) void gemm_nt_bias_bf16(
    const uint16_t* __restrict__ A, const uint16_t* __restrict__ B,
    const float* __restrict__ bias, uint16_t* __restrict__ C,
    int N, int Kd) {
  const int lane = threadIdx.x & 31;
  const int wid  = threadIdx.x >> 5;
  const int rowBase = blockIdx.y * 32;
  const int colBase = blockIdx.x * 256 + wid * 64;

  v8f acc[2][4] = {};
  for (int ks = 0; ks < Kd; ks += 32) {
    v16bf af[2], bfr[4];
#pragma unroll
    for (int mi = 0; mi < 2; ++mi)
      af[mi] = load_frag(A + (size_t)(rowBase + mi * 16) * Kd + ks, Kd, lane);
#pragma unroll
    for (int ni = 0; ni < 4; ++ni)
      bfr[ni] = load_frag(B + (size_t)(colBase + ni * 16) * Kd + ks, Kd, lane);
#pragma unroll
    for (int mi = 0; mi < 2; ++mi)
#pragma unroll
      for (int ni = 0; ni < 4; ++ni)
        acc[mi][ni] = wmma_bf16(af[mi], bfr[ni], acc[mi][ni]);
  }

  const int rlo = 8 * (lane >> 4);
#pragma unroll
  for (int ni = 0; ni < 4; ++ni) {
    const int col = colBase + ni * 16 + (lane & 15);
    const float bv = bias[col];
#pragma unroll
    for (int mi = 0; mi < 2; ++mi) {
#pragma unroll
      for (int j = 0; j < 8; ++j) {
        const int r = rowBase + mi * 16 + j + rlo;
        C[(size_t)r * N + col] = f2bf(acc[mi][ni][j] + bv);
      }
    }
  }
}

// ---------------------------------------------------------------------------
// 3) Gaussian blur of V along the sequence axis + transpose:
//    VbT[b][d][t] = sum_i w[i] * V[b][t+i-2][d]   (zero padded).
//    64x64 tile through LDS so both global read and write are coalesced.
// ---------------------------------------------------------------------------
__global__ __launch_bounds__(256) void blur_transpose(
    const uint16_t* __restrict__ V, uint16_t* __restrict__ VbT) {
  const float GW[5] = {0.054488685f, 0.244201342f, 0.402619946f,
                       0.244201342f, 0.054488685f};
  __shared__ uint16_t sIn[68][64];
  __shared__ uint16_t sT[64][65];

  const int b  = blockIdx.z;
  const int t0 = blockIdx.y * 64;
  const int d0 = blockIdx.x * 64;
  const uint16_t* Vb_ = V + (size_t)b * 1024 * 512;

  for (int idx = threadIdx.x; idx < 68 * 64; idx += 256) {
    const int rr = idx >> 6, cc = idx & 63;
    const int t = t0 - 2 + rr;
    uint16_t val = 0;
    if (t >= 0 && t < 1024) val = Vb_[(size_t)t * 512 + d0 + cc];
    sIn[rr][cc] = val;
  }
  __syncthreads();

  for (int idx = threadIdx.x; idx < 64 * 64; idx += 256) {
    const int tt = idx >> 6, dd = idx & 63;
    float acc = 0.0f;
#pragma unroll
    for (int i = 0; i < 5; ++i) acc += GW[i] * bf2f(sIn[tt + i][dd]);
    sT[dd][tt] = f2bf(acc);
  }
  __syncthreads();

  for (int idx = threadIdx.x; idx < 64 * 64; idx += 256) {
    const int r = idx >> 6, t = idx & 63;
    VbT[((size_t)b * 512 + d0 + r) * 1024 + t0 + t] = sT[r][t];
  }
}

// ---------------------------------------------------------------------------
// 4) Fused flash attention with pre-blurred V:
//    out = softmax(Q K^T / 16) @ Vblur.
//    128 threads = 4 waves per 16-query tile. Per 64-key block:
//      - wave w computes score tile [16 x 16] for keys kb+16w (8 WMMAs, H=256)
//      - parallel online softmax: 8 lanes per row, shfl-xor reductions
//      - P (16x64, bf16 in LDS) x Vblur (64 x 128 per wave): 16 WMMAs/wave
// ---------------------------------------------------------------------------
__global__ __launch_bounds__(128) void attn_blur_kernel(
    const uint16_t* __restrict__ Q, const uint16_t* __restrict__ Kmat,
    const uint16_t* __restrict__ VbT, float* __restrict__ out) {
  __shared__ float    sScores[16][64];
  __shared__ uint16_t sP[16][64];
  __shared__ float    sM[16], sL[16], sAlpha[16];

  const int tid  = threadIdx.x;
  const int lane = tid & 31;
  const int wid  = tid >> 5;
  const int b  = blockIdx.x >> 6;
  const int qt = blockIdx.x & 63;
  const size_t qRowG = (size_t)b * 1024 + qt * 16;
  const float scale = 0.0625f;  // 1/sqrt(H=256)

  v16bf qf[8];
#pragma unroll
  for (int s = 0; s < 8; ++s)
    qf[s] = load_frag(Q + qRowG * 256 + 32 * s, 256, lane);

  v8f oacc[8] = {};
  const int colBase = wid * 128;
  const int rlo = 8 * (lane >> 4);
  if (tid < 16) { sM[tid] = -__builtin_inff(); sL[tid] = 0.0f; }
  __syncthreads();

  for (int kb = 0; kb < 1024; kb += 64) {
    // ---- scores: this wave's 16x16 key tile over H=256 ----
    v8f sc = {};
    const size_t kRowG = (size_t)b * 1024 + kb + wid * 16;
#pragma unroll
    for (int s = 0; s < 8; ++s) {
      v16bf kf = load_frag(Kmat + kRowG * 256 + 32 * s, 256, lane);
      sc = wmma_bf16(qf[s], kf, sc);
    }
    // prefetch next key block (global_prefetch_b8) while WMMAs drain
    if (kb + 64 < 1024) {
      __builtin_prefetch(Kmat + (kRowG + 64) * 256 + (size_t)(lane & 15) * 256, 0, 1);
      __builtin_prefetch(VbT + ((size_t)b * 512 + colBase + (lane & 15)) * 1024 + kb + 64, 0, 1);
    }
#pragma unroll
    for (int j = 0; j < 8; ++j)
      sScores[j + rlo][wid * 16 + (lane & 15)] = sc[j] * scale;
    __syncthreads();

    // ---- parallel online softmax: 8 lanes per query row (same wave) ----
    {
      const int r  = tid >> 3;        // 0..15 (rows 0-3 in wave 0, etc.)
      const int cg = (tid & 7) * 8;   // this lane's 8-column group
      float vloc[8];
      float bmax = -__builtin_inff();
#pragma unroll
      for (int c = 0; c < 8; ++c) {
        vloc[c] = sScores[r][cg + c];
        bmax = fmaxf(bmax, vloc[c]);
      }
#pragma unroll
      for (int m = 4; m >= 1; m >>= 1)
        bmax = fmaxf(bmax, __shfl_xor(bmax, m, 8));
      const float m_old = sM[r];            // lockstep read before lane-0 write
      const float m_new = fmaxf(m_old, bmax);
      float sum = 0.0f;
#pragma unroll
      for (int c = 0; c < 8; ++c) {
        const float e = __expf(vloc[c] - m_new);
        sum += e;
        sP[r][cg + c] = f2bf(e);
      }
#pragma unroll
      for (int m = 4; m >= 1; m >>= 1)
        sum += __shfl_xor(sum, m, 8);
      if ((tid & 7) == 0) {
        const float alpha = __expf(m_old - m_new);
        sM[r] = m_new;
        sL[r] = sL[r] * alpha + sum;
        sAlpha[r] = alpha;
      }
    }
    __syncthreads();

    // ---- rescale running accumulators ----
    float al[8];
#pragma unroll
    for (int j = 0; j < 8; ++j) al[j] = sAlpha[j + rlo];
#pragma unroll
    for (int ct = 0; ct < 8; ++ct)
#pragma unroll
      for (int j = 0; j < 8; ++j) oacc[ct][j] *= al[j];

    // ---- P x Vblur : A-fragments from LDS, B-fragments from VbT ----
    v16bf pf0 = load_frag(&sP[0][0], 64, lane);       // K = kb+0..31
    v16bf pf1 = load_frag(&sP[0][0] + 32, 64, lane);  // K = kb+32..63
#pragma unroll
    for (int ct = 0; ct < 8; ++ct) {
      const size_t vRow = (size_t)b * 512 + colBase + ct * 16;
      v16bf v0 = load_frag(VbT + vRow * 1024 + kb, 1024, lane);
      v16bf v1 = load_frag(VbT + vRow * 1024 + kb + 32, 1024, lane);
      oacc[ct] = wmma_bf16(pf0, v0, oacc[ct]);
      oacc[ct] = wmma_bf16(pf1, v1, oacc[ct]);
    }
    __syncthreads();
  }

  // ---- normalize by l and store fp32 ----
  float linv[8];
#pragma unroll
  for (int j = 0; j < 8; ++j) linv[j] = 1.0f / sL[j + rlo];
#pragma unroll
  for (int ct = 0; ct < 8; ++ct) {
    const int col = colBase + ct * 16 + (lane & 15);
#pragma unroll
    for (int j = 0; j < 8; ++j) {
      const size_t r = qRowG + j + rlo;
      out[r * 512 + col] = oacc[ct][j] * linv[j];
    }
  }
}

// ---------------------------------------------------------------------------
// launch
// ---------------------------------------------------------------------------
extern "C" void kernel_launch(void* const* d_in, const int* in_sizes, int n_in,
                              void* d_out, int out_size, void* d_ws, size_t ws_size,
                              hipStream_t stream) {
  const float* x  = (const float*)d_in[0];
  const float* Wq = (const float*)d_in[1];
  const float* bq = (const float*)d_in[2];
  const float* Wk = (const float*)d_in[3];
  const float* bk = (const float*)d_in[4];
  const float* Wv = (const float*)d_in[5];
  const float* bv = (const float*)d_in[6];
  float* out = (float*)d_out;

  const int BC = 16 * 4, S = 1024, D = 512, H = 256;
  const size_t M = (size_t)BC * S;  // 65536 rows

  uint16_t* ws = (uint16_t*)d_ws;
  size_t off = 0;
  uint16_t* Xb  = ws + off; off += M * D;            // x in bf16
  uint16_t* Wqb = ws + off; off += (size_t)H * D;
  uint16_t* Wkb = ws + off; off += (size_t)H * D;
  uint16_t* Wvb = ws + off; off += (size_t)D * D;
  uint16_t* Qm  = ws + off; off += M * H;            // Q bf16
  uint16_t* Km  = ws + off; off += M * H;            // K bf16
  uint16_t* Vm  = ws + off; off += M * D;            // V bf16
  uint16_t* VbT = ws + off; off += M * D;            // blurred V, transposed
  (void)ws_size; (void)in_sizes; (void)n_in; (void)out_size;

  // 1) convert to bf16
  {
    int n8 = (int)(M * D / 8);
    cvt_f32_bf16<<<(n8 + 255) / 256, 256, 0, stream>>>(x, Xb, n8);
    n8 = (int)((size_t)H * D / 8);
    cvt_f32_bf16<<<(n8 + 255) / 256, 256, 0, stream>>>(Wq, Wqb, n8);
    cvt_f32_bf16<<<(n8 + 255) / 256, 256, 0, stream>>>(Wk, Wkb, n8);
    n8 = (int)((size_t)D * D / 8);
    cvt_f32_bf16<<<(n8 + 255) / 256, 256, 0, stream>>>(Wv, Wvb, n8);
  }

  // 2) QKV projections (WMMA bf16 GEMMs, fused bias)
  gemm_nt_bias_bf16<<<dim3(H / 256, (int)(M / 32)), 128, 0, stream>>>(
      Xb, Wqb, bq, Qm, H, D);
  gemm_nt_bias_bf16<<<dim3(H / 256, (int)(M / 32)), 128, 0, stream>>>(
      Xb, Wkb, bk, Km, H, D);
  gemm_nt_bias_bf16<<<dim3(D / 256, (int)(M / 32)), 128, 0, stream>>>(
      Xb, Wvb, bv, Vm, D, D);

  // 3) blur V along sequence axis into transposed layout
  blur_transpose<<<dim3(D / 64, S / 64, BC), 256, 0, stream>>>(Vm, VbT);

  // 4) fused flash attention (blur folded into V)
  attn_blur_kernel<<<BC * (S / 16), 128, 0, stream>>>(Qm, Km, VbT, out);
}
